// LightGCN_27384711480190
// MI455X (gfx1250) — matrix operand order repaired and checked by reading, not compiled, not run on hardware
//
#include <hip/hip_runtime.h>

#define NU_  100000
#define NI_  50000
#define NTOT 150000
#define DD   64
#define NB   2048

typedef __attribute__((ext_vector_type(16))) _Float16 v16h;
typedef __attribute__((ext_vector_type(8)))  _Float16 v8h;
typedef __attribute__((ext_vector_type(8)))  float    v8f;

#if defined(__HIP_DEVICE_COMPILE__) && defined(__AMDGCN__) && \
    __has_builtin(__builtin_amdgcn_global_load_async_to_lds_b128) && \
    __has_builtin(__builtin_amdgcn_s_wait_asynccnt)
#define USE_ASYNC_LDS 1
typedef int async_v4i __attribute__((vector_size(16)));            // matches builtin pointee
typedef __attribute__((address_space(1))) async_v4i gas_v4i;       // global (prints __device__)
typedef __attribute__((address_space(3))) async_v4i las_v4i;       // LDS
#else
#define USE_ASYNC_LDS 0
#endif

// ---------------- elementwise helpers ----------------

__global__ void zero_f4_kernel(float4* __restrict__ p, int n4) {
    int i = blockIdx.x * blockDim.x + threadIdx.x;
    if (i < n4) p[i] = make_float4(0.f, 0.f, 0.f, 0.f);
}

// out = a*in + b*out   (vectorized float4)
__global__ void axpby_f4_kernel(const float4* __restrict__ in, float4* __restrict__ out,
                                float a, float b, int n4) {
    int i = blockIdx.x * blockDim.x + threadIdx.x;
    if (i >= n4) return;
    float4 x = in[i];
    float4 y = out[i];
    y.x = a * x.x + b * y.x;
    y.y = a * x.y + b * y.y;
    y.z = a * x.z + b * y.z;
    y.w = a * x.w + b * y.w;
    out[i] = y;
}

// ---------------- spmm: side = A @ all_emb ----------------
// one wave32 per nnz; lane handles dims d and d+32. Atomic f32 scatter-add
// (global_atomic_add_f32); gather/scatter L2-resident (38MB table << 192MB L2).

__global__ void spmm_side_kernel(const float* __restrict__ user_emb,
                                 const float* __restrict__ item_emb,
                                 const float* __restrict__ vals,
                                 const int*   __restrict__ rows,
                                 const int*   __restrict__ cols,
                                 float* __restrict__ side, int nnz) {
    int lane = threadIdx.x & 31;
    int k = blockIdx.x * (blockDim.x >> 5) + (threadIdx.x >> 5);
    if (k >= nnz) return;
    if (lane == 0 && k + 8192 < nnz) {          // stream-ahead of index arrays
        __builtin_prefetch(rows + k + 8192, 0, 1);
        __builtin_prefetch(cols + k + 8192, 0, 1);
        __builtin_prefetch(vals + k + 8192, 0, 1);
    }
    int r = rows[k], c = cols[k];
    float v = vals[k];
    const float* src = (c < NU_) ? (user_emb + (size_t)c * DD)
                                 : (item_emb + (size_t)(c - NU_) * DD);
    float x0 = src[lane];
    float x1 = src[lane + 32];
    float* dst = side + (size_t)r * DD;
    atomicAdd(dst + lane,      v * x0);
    atomicAdd(dst + lane + 32, v * x1);
}

// ---------------- per-group masked spmm ----------------

__global__ void spmm_group_kernel(const float* __restrict__ src,
                                  const float* __restrict__ vals,
                                  const int*   __restrict__ rows,
                                  const int*   __restrict__ cols,
                                  const unsigned int* __restrict__ maskbuf,
                                  float* __restrict__ dst, int nnz, int g) {
    int lane = threadIdx.x & 31;
    int k = blockIdx.x * (blockDim.x >> 5) + (threadIdx.x >> 5);
    if (k >= nnz) return;
    if (lane == 0 && k + 8192 < nnz) {
        __builtin_prefetch(rows + k + 8192, 0, 1);
        __builtin_prefetch(cols + k + 8192, 0, 1);
        __builtin_prefetch(vals + k + 8192, 0, 1);
    }
    int r = rows[k], c = cols[k];
    unsigned int m = maskbuf[r] & maskbuf[c];   // wave-uniform -> cheap skip
    if (!((m >> g) & 1u)) return;
    float v = vals[k];
    const float* s = src + (size_t)c * DD;
    float* d = dst + (size_t)r * DD;
    atomicAdd(d + lane,      v * s[lane]);
    atomicAdd(d + lane + 32, v * s[lane + 32]);
}

// ---------------- fused fc -> leaky -> fc_g -> one-hot mask (WMMA) ----------------
// Block = 128 threads = 4 wave32, 64 rows of x = all_emb + side per block.
// fc_w tile staged into LDS via async LDS-DMA (GLOBAL_LOAD_ASYNC_TO_LDS_B128,
// s_wait_asynccnt) overlapping the x-tile gather, then transposed to whT so
// every WMMA operand fragment is two aligned ds_load_b128 per lane:
//   A (16x32 f16): lane needs K = 32s+kofs_a..+7 and 32s+16+kofs_a..+7 at row lane%16
//   B (32x16 f16): lane needs K = 32s+kofs_b..+7 and 32s+16+kofs_b..+7 at column lane%16

__global__ void __launch_bounds__(128)
fc_oh_kernel(const float* __restrict__ user_emb, const float* __restrict__ item_emb,
             const float* __restrict__ side,
             const float* __restrict__ fc_w, const float* __restrict__ fc_b,
             const float* __restrict__ fcg_w, const float* __restrict__ fcg_b,
             unsigned int* __restrict__ maskbuf) {
    __shared__ __align__(16) _Float16 xh[64][72];   // x tile, row-major (K contiguous)
    __shared__ __align__(16) _Float16 whT[64][72];  // fc_w transposed: whT[j][k]
    __shared__ __align__(16) float    ytile[64][68]; // staging for fc_w, then results
    __shared__ float s_fcb[64];
    __shared__ float s_fcgw[DD * 4];
    __shared__ float s_fcgb[4];

    const int tid  = threadIdx.x;
    const int row0 = blockIdx.x * 64;
    float* stage = &ytile[0][0];                    // 4096 floats used for staging

#if USE_ASYNC_LDS
    {   // async DMA: fc_w (16KB) -> LDS staging; 16B per lane per issue
        gas_v4i* gsrc = (gas_v4i*)fc_w;
        las_v4i* ldst = (las_v4i*)stage;
        for (int i = tid; i < 1024; i += 128)
            __builtin_amdgcn_global_load_async_to_lds_b128(gsrc + i, ldst + i, 0, 0);
    }
#else
    for (int i = tid; i < 4096; i += 128) stage[i] = fc_w[i];
#endif

    // x tile gather (overlaps with the async weight DMA)
    for (int i = tid; i < 64 * 64; i += 128) {
        int r = i >> 6, d = i & 63;
        int n = row0 + r;
        float v = 0.f;
        if (n < NTOT) {
            float e = (n < NU_) ? user_emb[(size_t)n * DD + d]
                                : item_emb[(size_t)(n - NU_) * DD + d];
            v = e + side[(size_t)n * DD + d];
        }
        xh[r][d] = (_Float16)v;
    }
    if (tid < 64) s_fcb[tid] = fc_b[tid];
    for (int i = tid; i < DD * 4; i += 128) s_fcgw[i] = fcg_w[i];
    if (tid < 4) s_fcgb[tid] = fcg_b[tid];

#if USE_ASYNC_LDS
    __builtin_amdgcn_s_wait_asynccnt(0);
#endif
    __syncthreads();

    // transpose-convert weights: whT[j][k] = (f16) fc_w[k*64 + j]
    for (int i = tid; i < 4096; i += 128) {
        int k = i >> 6, j = i & 63;
        whT[j][k] = (_Float16)stage[i];
    }
    __syncthreads();    // whT ready; ytile(staging) now dead -> reusable for results

    const int wave   = tid >> 5;
    const int lane   = tid & 31;
    const int lm     = lane & 15;
    const int mrow   = wave * 16;            // 16 rows per wave
    const int kofs_a = (lane < 16) ? 0 : 8;  // A lane-half K offset
    const int kofs_b = (lane < 16) ? 0 : 8;  // B lane-half K offset
    const int rbase  = mrow + ((lane < 16) ? 0 : 8);  // C/D: M = v + 8*(lane>=16)

    for (int t = 0; t < 4; ++t) {            // 4 n-tiles of 16 cols
        v8f c = {0.f, 0.f, 0.f, 0.f, 0.f, 0.f, 0.f, 0.f};
        const int ncol = t * 16 + lm;        // B/D: N = lane%16
#pragma unroll
        for (int s = 0; s < 2; ++s) {        // K = 64 in 2 steps of 32
            v8h a_lo = *(const v8h*)&xh[mrow + lm][32 * s + kofs_a];
            v8h a_hi = *(const v8h*)&xh[mrow + lm][32 * s + 16 + kofs_a];
            v8h b_lo = *(const v8h*)&whT[ncol][32 * s + kofs_b];
            v8h b_hi = *(const v8h*)&whT[ncol][32 * s + 16 + kofs_b];
            v16h a = __builtin_shufflevector(a_lo, a_hi,
                        0, 1, 2, 3, 4, 5, 6, 7, 8, 9, 10, 11, 12, 13, 14, 15);
            v16h b = __builtin_shufflevector(b_lo, b_hi,
                        0, 1, 2, 3, 4, 5, 6, 7, 8, 9, 10, 11, 12, 13, 14, 15);
            c = __builtin_amdgcn_wmma_f32_16x16x32_f16(
                    false, a, false, b, (short)0, c, false, false);
        }
#pragma unroll
        for (int v = 0; v < 8; ++v) {
            float y = c[v] + s_fcb[ncol];
            y = (y >= 0.f) ? y : 0.01f * y;              // leaky_relu(0.01)
            ytile[rbase + v][ncol] = y;
        }
    }
    __syncthreads();

    if (tid < 64) {
        int n = row0 + tid;
        if (n < NTOT) {
            unsigned int m;
            if (n >= NU_) {
                m = 0xFu;                                // items: all-ones rows
            } else {
                float s0 = s_fcgb[0], s1 = s_fcgb[1], s2 = s_fcgb[2], s3 = s_fcgb[3];
#pragma unroll 8
                for (int d = 0; d < DD; ++d) {
                    float x = ytile[tid][d];
                    s0 += x * s_fcgw[d * 4 + 0];
                    s1 += x * s_fcgw[d * 4 + 1];
                    s2 += x * s_fcgw[d * 4 + 2];
                    s3 += x * s_fcgw[d * 4 + 3];
                }
                float mx = fmaxf(fmaxf(s0, s1), fmaxf(s2, s3));
                m = (unsigned)(s0 == mx) | ((unsigned)(s1 == mx) << 1)
                  | ((unsigned)(s2 == mx) << 2) | ((unsigned)(s3 == mx) << 3);
            }
            maskbuf[n] = m;
        }
    }
}

// ---------------- gamma: per-pair dot of final embeddings ----------------
// final = 0.2*acc on both sides -> factor 0.04. One wave32 per pair.

__global__ void gamma_kernel(const float* __restrict__ acc,
                             const int* __restrict__ users,
                             const int* __restrict__ items,
                             float* __restrict__ out, int nb) {
    int lane = threadIdx.x & 31;
    int b = blockIdx.x * (blockDim.x >> 5) + (threadIdx.x >> 5);
    if (b >= nb) return;
    const float* pu = acc + (size_t)users[b] * DD;
    const float* pi = acc + (size_t)(NU_ + items[b]) * DD;
    float s = pu[lane] * pi[lane] + pu[lane + 32] * pi[lane + 32];
#pragma unroll
    for (int off = 16; off > 0; off >>= 1) s += __shfl_down(s, off, 32);
    if (lane == 0) out[b] = 0.04f * s;
}

// ---------------- driver ----------------

extern "C" void kernel_launch(void* const* d_in, const int* in_sizes, int n_in,
                              void* d_out, int out_size, void* d_ws, size_t ws_size,
                              hipStream_t stream) {
    const float* user_emb = (const float*)d_in[0];
    const float* item_emb = (const float*)d_in[1];
    const float* fc_w     = (const float*)d_in[2];
    const float* fc_b     = (const float*)d_in[3];
    const float* fcg_w    = (const float*)d_in[4];
    const float* fcg_b    = (const float*)d_in[5];
    const float* vals     = (const float*)d_in[6];
    const int*   rows     = (const int*)  d_in[7];
    const int*   cols     = (const int*)  d_in[8];
    const int*   users    = (const int*)  d_in[9];
    const int*   items    = (const int*)  d_in[10];
    float*       out      = (float*)d_out;

    const int nnz = in_sizes[6];
    const int nb  = in_sizes[9];

    const size_t NE = (size_t)NTOT * DD;           // 9,600,000 floats
    float* side = (float*)d_ws;
    float* acc  = side + NE;
    float* e0   = acc + NE;
    float* e1   = e0 + NE;
    unsigned int* maskbuf = (unsigned int*)(e1 + NE);

    const int n4  = (int)(NE / 4);
    const int eb  = 256;
    const int eg  = (n4 + eb - 1) / eb;
    const int sg  = (nnz + 7) / 8;                 // 8 nnz per 256-thread block
    const int fcg = (NTOT + 63) / 64;

    // side = spmm(vals, all_emb)
    zero_f4_kernel<<<eg, eb, 0, stream>>>((float4*)side, n4);
    spmm_side_kernel<<<sg, 256, 0, stream>>>(user_emb, item_emb, vals, rows, cols, side, nnz);

    // one-hot group masks (WMMA dense fc)
    fc_oh_kernel<<<fcg, 128, 0, stream>>>(user_emb, item_emb, side,
                                          fc_w, fc_b, fcg_w, fcg_b, maskbuf);

    // layer 0: acc = G * side
    axpby_f4_kernel<<<eg, eb, 0, stream>>>((const float4*)side, (float4*)acc, 4.f, 0.f, n4);

    // layers 1..2, serial over groups (ping-pong e0/e1)
    for (int g = 0; g < 4; ++g) {
        zero_f4_kernel<<<eg, eb, 0, stream>>>((float4*)e0, n4);
        spmm_group_kernel<<<sg, 256, 0, stream>>>(side, vals, rows, cols, maskbuf, e0, nnz, g);
        axpby_f4_kernel<<<eg, eb, 0, stream>>>((const float4*)e0, (float4*)acc, 1.f, 1.f, n4);

        zero_f4_kernel<<<eg, eb, 0, stream>>>((float4*)e1, n4);
        spmm_group_kernel<<<sg, 256, 0, stream>>>(e0, vals, rows, cols, maskbuf, e1, nnz, g);
        axpby_f4_kernel<<<eg, eb, 0, stream>>>((const float4*)e1, (float4*)acc, 1.f, 1.f, n4);
    }

    // gamma = 0.04 * <acc_u, acc_i>
    gamma_kernel<<<(nb + 7) / 8, 256, 0, stream>>>(acc, users, items, out, nb);
}
